// DEMONetHashGraph_3083786518802
// MI455X (gfx1250) — matrix operand order repaired and checked by reading, not compiled, not run on hardware
//
#include <hip/hip_runtime.h>
#include <hip/hip_bf16.h>
#include <stdint.h>

#define N_NODES   50000
#define N_EDGES   800000
#define DMODEL    512
#define KDIM      1024          // concat [agg ; h]
#define NUM_HASH  4
#define HASH_DIM  128
#define NUM_GRAPHS 64
#define NUM_CLASSES 10

typedef __bf16 bf16;
typedef __attribute__((ext_vector_type(16))) __bf16 v16bf;
typedef __attribute__((ext_vector_type(8)))  __bf16 v8bf;
typedef __attribute__((ext_vector_type(8)))  float  v8f;

union FragBF { v16bf v; v8bf h[2]; };

// ---------------- CDNA5 async global->LDS helpers ----------------
__device__ __forceinline__ void async_b128(const void* lds_ptr, const void* gptr) {
    // LDS byte offset = low 32 bits of generic shared address (aperture rule).
    unsigned ldsoff = (unsigned)(uintptr_t)lds_ptr;
    asm volatile("global_load_async_to_lds_b128 %0, %1, off"
                 :: "v"(ldsoff), "v"(gptr) : "memory");
}
__device__ __forceinline__ void wait_async0() {
    asm volatile("s_wait_asynccnt 0" ::: "memory");
}

// ---------------- utility kernels ----------------
__global__ void k_fill0(float* __restrict__ p, int n) {
    int i = blockIdx.x * blockDim.x + threadIdx.x;
    if (i < n) p[i] = 0.0f;
}

__global__ void k_deg(const long long* __restrict__ ei, float* __restrict__ deg) {
    int e = blockIdx.x * blockDim.x + threadIdx.x;
    if (e < N_EDGES) atomicAdd(&deg[(int)ei[e]], 1.0f);
}

// nsum[src] += h[dst], 4 floats per thread
__global__ void k_scatter(const long long* __restrict__ ei,
                          const float* __restrict__ h,
                          float* __restrict__ nsum) {
    int tid = blockIdx.x * blockDim.x + threadIdx.x;
    if (tid >= N_EDGES * 128) return;
    int e = tid >> 7;
    int c = (tid & 127) << 2;
    int s = (int)ei[e];
    int d = (int)ei[N_EDGES + e];
    const float4 v = *(const float4*)(h + (size_t)d * DMODEL + c);
    float* p = nsum + (size_t)s * DMODEL + c;
    atomicAdd(p + 0, v.x); atomicAdd(p + 1, v.y);
    atomicAdd(p + 2, v.z); atomicAdd(p + 3, v.w);
}

// WT[j][kk] : kk<512 -> (Hm-folded Wp) row kk col j ; kk>=512 -> Ws[kk-512][j]
__global__ void k_build_wcat(const float* __restrict__ Hm,
                             const float* __restrict__ Wp,
                             const float* __restrict__ Ws,
                             bf16* __restrict__ WT) {
    int tid = blockIdx.x * blockDim.x + threadIdx.x;
    if (tid >= DMODEL * KDIM) return;
    int j  = tid >> 10;
    int kk = tid & (KDIM - 1);
    float acc;
    if (kk < DMODEL) {
        acc = 0.0f;
        for (int k = 0; k < NUM_HASH; ++k) {
            const float* hm = Hm + ((size_t)k * DMODEL + kk) * HASH_DIM;
            const float* wp = Wp + (size_t)k * HASH_DIM * DMODEL + j;
            for (int hh = 0; hh < HASH_DIM; ++hh)
                acc += hm[hh] * wp[(size_t)hh * DMODEL];
        }
    } else {
        acc = Ws[(size_t)(kk - DMODEL) * DMODEL + j];
    }
    WT[(size_t)j * KDIM + kk] = (bf16)acc;
}

// X[n][0:512] = bf16(agg), X[n][512:1024] = bf16(h)
__global__ void k_aggcvt(const float* __restrict__ h,
                         const float* __restrict__ nsum,
                         const float* __restrict__ deg,
                         bf16* __restrict__ X) {
    int tid = blockIdx.x * blockDim.x + threadIdx.x;
    if (tid >= N_NODES * DMODEL) return;
    int n = tid >> 9, j = tid & (DMODEL - 1);
    float dg = deg[n];
    float hv = h[(size_t)n * DMODEL + j];
    float a  = (dg > 0.0f) ? nsum[(size_t)n * DMODEL + j] / dg : hv;
    X[(size_t)n * KDIM + j]          = (bf16)a;
    X[(size_t)n * KDIM + DMODEL + j] = (bf16)hv;
}

// ---------------- WMMA bf16 GEMM: out = elu(X[N,1024] @ W[1024,512] + b) ----------------
// Double-buffered async global->LDS pipeline: loads for step s+1 are in flight
// while step s computes its 8 WMMAs.
#define TILE  128
#define LDSTR 40   // 32 bf16 K-slice padded to 40 elems (80B rows, conflict-free)
#define KSTEPS (KDIM / 32)

__global__ __launch_bounds__(256) void k_gemm_wmma(
        const bf16* __restrict__ X,    // [nRows][KDIM]
        const bf16* __restrict__ WT,   // [DMODEL][KDIM]  (transposed weight)
        const float* __restrict__ bias,// [DMODEL]
        float* __restrict__ out,       // [nRows][DMODEL]
        int nRows) {
    __shared__ __align__(16) bf16 As[2][TILE * LDSTR];
    __shared__ __align__(16) bf16 Bs[2][TILE * LDSTR];

    const int t    = threadIdx.x;
    const int lane = t & 31;
    const int wave = t >> 5;
    const int m0   = blockIdx.x * TILE;
    const int n0   = blockIdx.y * TILE;

    // per-thread staging coordinates: 2 chunks of 16B for A, 2 for B
    int srow[2], ssub[2];
    const bf16* gA[2];
    const bf16* gB[2];
#pragma unroll
    for (int i = 0; i < 2; ++i) {
        int c   = t + 256 * i;
        srow[i] = c >> 2;
        ssub[i] = c & 3;
        int gr  = m0 + srow[i]; if (gr >= nRows) gr = nRows - 1;     // clamp (safe dup)
        gA[i] = X  + (size_t)gr * KDIM + ssub[i] * 8;
        gB[i] = WT + (size_t)(n0 + srow[i]) * KDIM + ssub[i] * 8;
    }

    auto stage = [&](int buf, int kb) {
#pragma unroll
        for (int i = 0; i < 2; ++i) {
            async_b128(&As[buf][srow[i] * LDSTR + ssub[i] * 8], gA[i] + kb);
            async_b128(&Bs[buf][srow[i] * LDSTR + ssub[i] * 8], gB[i] + kb);
        }
    };

    v8f acc[8];
#pragma unroll
    for (int j = 0; j < 8; ++j) acc[j] = 0.0f;

    const int arow = wave * 16 + (lane & 15);     // A fragment row within tile
    const int kg   = lane >> 4;                   // K-group (lo/hi lane half)

    stage(0, 0);                                  // prologue
    for (int s = 0; s < KSTEPS; ++s) {
        wait_async0();        // this wave's buffer-s loads complete
        __syncthreads();      // everyone's data visible; everyone done with buf s^1
        if (s + 1 < KSTEPS)
            stage((s + 1) & 1, (s + 1) * 32);     // prefetch next tile under compute
        const bf16* as = As[s & 1];
        const bf16* bs = Bs[s & 1];

        // A fragment (16x32 bf16): elems 0-7 = K(kg*8 ..), elems 8-15 = K(16+kg*8 ..)
        FragBF a;
        a.h[0] = *(const v8bf*)(&as[arow * LDSTR + kg * 8]);
        a.h[1] = *(const v8bf*)(&as[arow * LDSTR + 16 + kg * 8]);

#pragma unroll
        for (int j = 0; j < 8; ++j) {
            const int bcol = j * 16 + (lane & 15);
            FragBF b;       // B fragment (32x16): lane col, elems 0-7 = K(kg*8 ..), elems 8-15 = K(16+kg*8 ..)
            b.h[0] = *(const v8bf*)(&bs[bcol * LDSTR + kg * 8]);
            b.h[1] = *(const v8bf*)(&bs[bcol * LDSTR + 16 + kg * 8]);
            acc[j] = __builtin_amdgcn_wmma_f32_16x16x32_bf16(
                         false, a.v, false, b.v, (short)0, acc[j], false, false);
        }
    }

    // epilogue: C/D layout -> row = vgpr + (lane>>4)*8, col = lane&15
    const int colbase = n0 + (lane & 15);
    const int rowbase = m0 + wave * 16 + (lane >> 4) * 8;
#pragma unroll
    for (int j = 0; j < 8; ++j) {
        const int col = colbase + j * 16;
        const float bv = bias[col];
#pragma unroll
        for (int v = 0; v < 8; ++v) {
            const int r = rowbase + v;
            if (r < nRows) {
                float x = acc[j][v] + bv;
                out[(size_t)r * DMODEL + col] = (x > 0.0f) ? x : (__expf(x) - 1.0f);
            }
        }
    }
}

// ---------------- pooling + classifier ----------------
__global__ void k_gcnt(const long long* __restrict__ batch, float* __restrict__ gcnt) {
    int n = blockIdx.x * blockDim.x + threadIdx.x;
    if (n < N_NODES) atomicAdd(&gcnt[(int)batch[n]], 1.0f);
}

__global__ void k_pool(const long long* __restrict__ batch,
                       const float* __restrict__ h, float* __restrict__ gsum) {
    int tid = blockIdx.x * blockDim.x + threadIdx.x;
    if (tid >= N_NODES * 128) return;
    int n = tid >> 7;
    int c = (tid & 127) << 2;
    int b = (int)batch[n];
    const float4 v = *(const float4*)(h + (size_t)n * DMODEL + c);
    float* p = gsum + (size_t)b * DMODEL + c;
    atomicAdd(p + 0, v.x); atomicAdd(p + 1, v.y);
    atomicAdd(p + 2, v.z); atomicAdd(p + 3, v.w);
}

__global__ void k_cls(const float* __restrict__ gsum, const float* __restrict__ gcnt,
                      const float* __restrict__ Wc, const float* __restrict__ bc,
                      float* __restrict__ out) {
    int tid = blockIdx.x * blockDim.x + threadIdx.x;
    if (tid >= NUM_GRAPHS * NUM_CLASSES) return;
    int g = tid / NUM_CLASSES, c = tid % NUM_CLASSES;
    float s = 0.0f;
    for (int j = 0; j < DMODEL; ++j)
        s += gsum[(size_t)g * DMODEL + j] * Wc[(size_t)j * NUM_CLASSES + c];
    float cnt = gcnt[g]; if (cnt < 1.0f) cnt = 1.0f;
    out[tid] = s / cnt + bc[c];
}

// ---------------- host launcher ----------------
extern "C" void kernel_launch(void* const* d_in, const int* in_sizes, int n_in,
                              void* d_out, int out_size, void* d_ws, size_t ws_size,
                              hipStream_t stream) {
    const float*     x     = (const float*)d_in[0];
    const long long* ei    = (const long long*)d_in[1];
    const long long* batch = (const long long*)d_in[2];
    const float* Hm1 = (const float*)d_in[3];
    const float* Wp1 = (const float*)d_in[4];
    const float* Ws1 = (const float*)d_in[5];
    const float* b1  = (const float*)d_in[6];
    const float* Hm2 = (const float*)d_in[7];
    const float* Wp2 = (const float*)d_in[8];
    const float* Ws2 = (const float*)d_in[9];
    const float* b2  = (const float*)d_in[10];
    const float* Wc  = (const float*)d_in[11];
    const float* bc  = (const float*)d_in[12];
    float* out = (float*)d_out;

    char* ws = (char*)d_ws;
    size_t off = 0;
    auto carve = [&](size_t bytes) { void* p = ws + off; off += (bytes + 255) & ~(size_t)255; return p; };
    float* nsum = (float*)carve((size_t)N_NODES * DMODEL * 4);
    bf16*  Xcat = (bf16*) carve((size_t)N_NODES * KDIM * 2);
    float* hbuf = (float*)carve((size_t)N_NODES * DMODEL * 4);
    bf16*  WT   = (bf16*) carve((size_t)DMODEL * KDIM * 2);
    float* deg  = (float*)carve((size_t)N_NODES * 4);
    float* gsum = (float*)carve((size_t)NUM_GRAPHS * DMODEL * 4);
    float* gcnt = (float*)carve((size_t)NUM_GRAPHS * 4);

    const int B = 256;
    const int nElemNode = N_NODES * DMODEL;           // 25.6M
    const int nElemEdge = N_EDGES * 128;              // 102.4M
    dim3 ggrid((N_NODES + TILE - 1) / TILE, DMODEL / TILE);  // (391, 4)

    // degrees
    k_fill0<<<(N_NODES + B - 1) / B, B, 0, stream>>>(deg, N_NODES);
    k_deg<<<(N_EDGES + B - 1) / B, B, 0, stream>>>(ei, deg);

    // ---- layer 1 ----
    k_fill0<<<(nElemNode + B - 1) / B, B, 0, stream>>>(nsum, nElemNode);
    k_scatter<<<(nElemEdge + B - 1) / B, B, 0, stream>>>(ei, x, nsum);
    k_build_wcat<<<(DMODEL * KDIM + B - 1) / B, B, 0, stream>>>(Hm1, Wp1, Ws1, WT);
    k_aggcvt<<<(nElemNode + B - 1) / B, B, 0, stream>>>(x, nsum, deg, Xcat);
    k_gemm_wmma<<<ggrid, B, 0, stream>>>(Xcat, WT, b1, hbuf, N_NODES);

    // ---- layer 2 ----
    k_fill0<<<(nElemNode + B - 1) / B, B, 0, stream>>>(nsum, nElemNode);
    k_scatter<<<(nElemEdge + B - 1) / B, B, 0, stream>>>(ei, hbuf, nsum);
    k_build_wcat<<<(DMODEL * KDIM + B - 1) / B, B, 0, stream>>>(Hm2, Wp2, Ws2, WT);
    k_aggcvt<<<(nElemNode + B - 1) / B, B, 0, stream>>>(hbuf, nsum, deg, Xcat);
    k_gemm_wmma<<<ggrid, B, 0, stream>>>(Xcat, WT, b2, hbuf, N_NODES);

    // ---- pool + classify ----
    k_fill0<<<(NUM_GRAPHS * DMODEL + B - 1) / B, B, 0, stream>>>(gsum, NUM_GRAPHS * DMODEL);
    k_fill0<<<1, B, 0, stream>>>(gcnt, NUM_GRAPHS);
    k_gcnt<<<(N_NODES + B - 1) / B, B, 0, stream>>>(batch, gcnt);
    k_pool<<<(N_NODES * 128 + B - 1) / B, B, 0, stream>>>(batch, hbuf, gsum);
    k_cls<<<(NUM_GRAPHS * NUM_CLASSES + 63) / 64, 64, 0, stream>>>(gsum, gcnt, Wc, bc, out);
}
